// MoFo_Circulant_Attention_5437428597097
// MI455X (gfx1250) — matrix operand order repaired and checked by reading, not compiled, not run on hardware
//
#include <hip/hip_runtime.h>
#include <hip/hip_bf16.h>

typedef __bf16 bf16;
typedef __attribute__((ext_vector_type(16))) __bf16 v16bf;
typedef __attribute__((ext_vector_type(8)))  float  v8f;
typedef unsigned int u32x4 __attribute__((ext_vector_type(4)));

union Frag   { v16bf v; u32x4 q[2]; };
union Pack16 { u32x4 q[2]; bf16 h[16]; };

__device__ __forceinline__ v8f wmma_bf16(v16bf a, v16bf b, v8f c) {
  // (neg_a, A, neg_b, B, c_mod, C, reuse_a, reuse_b)
  return __builtin_amdgcn_wmma_f32_16x16x32_bf16(false, a, false, b, (short)0, c, false, false);
}

// ---------------------------------------------------------------------------
// f32 -> bf16 conversion (grid-stride)
// ---------------------------------------------------------------------------
__global__ void cvt_f32_bf16(const float* __restrict__ s, bf16* __restrict__ d, int n) {
  int i = blockIdx.x * blockDim.x + threadIdx.x;
  int stride = gridDim.x * blockDim.x;
  for (; i < n; i += stride) d[i] = (bf16)s[i];
}

// ---------------------------------------------------------------------------
// bf16 WMMA GEMM: C[M,N] = A[M,K] x B[K,N] + bias, fused epilogue.
// MODE 0: store bf16 (qkv).  MODE 1: SiLU, store f32 (gate).
// MODE 2: multiply by gate[], store f32 (final out-projection).
//
// Block: 256 threads = 8 waves. Macro tile 128(M) x 128(N), K-step 32.
// Wave w: rows (w&3)*32..+31 (2 m-tiles), cols (w>>2)*64..+63 (4 n-tiles)
// -> 8 WMMA accumulators, 8 WMMAs per K-stage, B fragments reused x2.
// Double-buffered LDS: one barrier per K-step; global loads for the next
// stage are issued before the WMMA burst so latency hides under compute.
// ---------------------------------------------------------------------------
template <int MODE>
__global__ __launch_bounds__(256)
void gemm_bf16_wmma(const bf16* __restrict__ A, const bf16* __restrict__ B,
                    const float* __restrict__ bias, const float* __restrict__ gate,
                    float* __restrict__ outF, bf16* __restrict__ outB,
                    int M, int N, int K)
{
  __shared__ bf16 As[2][128][32];   // [buf][m][k]
  __shared__ bf16 Bs[2][128][40];   // [buf][n][k], padded row (80B, 16B-aligned)

  const int tid  = threadIdx.x;
  const int lane = tid & 31;
  const int w    = tid >> 5;
  const int wm   = w & 3;          // 32-row group
  const int wn   = w >> 2;         // 64-col group
  const int mb   = blockIdx.y * 128;
  const int nb   = blockIdx.x * 128;
  const int l15  = lane & 15;
  const int hi   = lane >> 4;      // lane half selects K sub-range per ISA layout

  v8f acc[2][4] = {};

  const int arow = tid >> 1, ach = (tid & 1) * 16;  // A staging: 32B per thread
  const int bk   = tid >> 3, bn0 = (tid & 7) * 16;  // B staging: 32B, transposed

  const int KT = K >> 5;

  auto stage = [&](int buf, int k0) {
    const bf16* aptr = A + (size_t)(mb + arow) * K + k0 + ach;
    *(u32x4*)&As[buf][arow][ach]     = *(const u32x4*)(aptr);
    *(u32x4*)&As[buf][arow][ach + 8] = *(const u32x4*)(aptr + 8);
    Pack16 p;
    const bf16* bptr = B + (size_t)(k0 + bk) * N + nb + bn0;
    p.q[0] = *(const u32x4*)(bptr);
    p.q[1] = *(const u32x4*)(bptr + 8);
#pragma unroll
    for (int i = 0; i < 16; ++i) Bs[buf][bn0 + i][bk] = p.h[i];
  };

  stage(0, 0);
  int buf = 0;
  for (int kt = 0; kt < KT; ++kt) {
    __syncthreads();                       // stage(buf) visible; buf^1 free
    if (kt + 1 < KT) stage(buf ^ 1, (kt + 1) * 32);
    if (kt + 2 < KT) {                     // warm GL2 for tile after next
      __builtin_prefetch(A + (size_t)(mb + arow) * K + (kt + 2) * 32 + ach, 0, 1);
      __builtin_prefetch(B + (size_t)((kt + 2) * 32 + bk) * N + nb + bn0, 0, 1);
    }

    Frag af[2];
#pragma unroll
    for (int mi = 0; mi < 2; ++mi) {
      const int row = wm * 32 + mi * 16 + l15;
      af[mi].q[0] = *(const u32x4*)&As[buf][row][hi * 8];
      af[mi].q[1] = *(const u32x4*)&As[buf][row][hi * 8 + 16];
    }
    const int kb = hi * 8;
#pragma unroll
    for (int j = 0; j < 4; ++j) {
      Frag bfr;
      const int col = wn * 64 + j * 16 + l15;
      bfr.q[0] = *(const u32x4*)&Bs[buf][col][kb];
      bfr.q[1] = *(const u32x4*)&Bs[buf][col][kb + 16];
#pragma unroll
      for (int mi = 0; mi < 2; ++mi)
        acc[mi][j] = wmma_bf16(af[mi].v, bfr.v, acc[mi][j]);
    }
    buf ^= 1;
  }

  // epilogue: C element e <-> row e (lanes 0-15) / e+8 (lanes 16-31)
#pragma unroll
  for (int mi = 0; mi < 2; ++mi) {
    const int growb = mb + wm * 32 + mi * 16 + hi * 8;
#pragma unroll
    for (int j = 0; j < 4; ++j) {
      const int gcol = nb + wn * 64 + j * 16 + l15;
      const float bv = bias[gcol];
#pragma unroll
      for (int e = 0; e < 8; ++e) {
        const size_t idx = (size_t)(growb + e) * N + gcol;
        float v = acc[mi][j][e] + bv;
        if (MODE == 0) {
          outB[idx] = (bf16)v;
        } else if (MODE == 1) {
          outF[idx] = v / (1.0f + __expf(-v));     // SiLU
        } else {
          outF[idx] = v * gate[idx];               // token reweight
        }
      }
    }
  }
}

// ---------------------------------------------------------------------------
// Flash attention over qkv [8192, 3072] bf16 (layout [B,T,H, q|k|v of 64]).
// Grid: B*H*(T/128) blocks; block = 256 threads = 8 waves; wave owns 16 query
// rows -> softmax stats are wave-local (shfl within 16-lane halves).
// ---------------------------------------------------------------------------
__global__ __launch_bounds__(256)
void attention_kernel(const bf16* __restrict__ qkv, bf16* __restrict__ attn_out)
{
  __shared__ bf16 Qs[128][64];      // [row][hd]
  __shared__ bf16 Ks[64][64];       // [key][hd]
  __shared__ bf16 Vts[64][64];      // [hd][key] (transposed)
  __shared__ bf16 Ps[8][16][64];    // per-wave P tile [row][key]

  const int T = 1024, C3 = 3072, D = 1024;
  const int tid = threadIdx.x, lane = tid & 31, w = tid >> 5;
  const int l15 = lane & 15, hi = lane >> 4;

  const int qblk = blockIdx.x & 7;
  const int bh   = blockIdx.x >> 3;
  const int h    = bh & 15, b = bh >> 4;

  // stage Q (wave-local rows: tid>>1 maps wave w to rows 16w..16w+15)
  {
    const int row = tid >> 1, half = (tid & 1) * 32;
    const bf16* src = qkv + (size_t)(b * T + qblk * 128 + row) * C3 + h * 192 + half;
#pragma unroll
    for (int i = 0; i < 4; ++i)
      *(u32x4*)&Qs[row][half + i * 8] = *(const u32x4*)(src + i * 8);
  }

  v8f o[4] = {};
  float mrow[8], lrow[8];
#pragma unroll
  for (int e = 0; e < 8; ++e) { mrow[e] = -1e30f; lrow[e] = 0.f; }
  const float scale = 0.125f;   // 64^-0.5

  for (int kb0 = 0; kb0 < T; kb0 += 64) {
    __syncthreads();            // all waves done reading previous Ks/Vts
    {
      const int kr = tid >> 2, c = (tid & 3) * 16;
      const bf16* ksrc = qkv + (size_t)(b * T + kb0 + kr) * C3 + h * 192 + 64 + c;
      *(u32x4*)&Ks[kr][c]     = *(const u32x4*)(ksrc);
      *(u32x4*)&Ks[kr][c + 8] = *(const u32x4*)(ksrc + 8);
      Pack16 pv;
      const bf16* vsrc = qkv + (size_t)(b * T + kb0 + kr) * C3 + h * 192 + 128 + c;
      pv.q[0] = *(const u32x4*)(vsrc);
      pv.q[1] = *(const u32x4*)(vsrc + 8);
#pragma unroll
      for (int i = 0; i < 16; ++i) Vts[c + i][kr] = pv.h[i];
    }
    __syncthreads();

    // S = Q K^T  (contraction over hd=64, two K=32 steps)
    v8f s[4] = {};
#pragma unroll
    for (int ks = 0; ks < 2; ++ks) {
      Frag af;
      const int abase = ks * 32 + hi * 8;
      af.q[0] = *(const u32x4*)&Qs[w * 16 + l15][abase];
      af.q[1] = *(const u32x4*)&Qs[w * 16 + l15][abase + 16];
      const int kbb = ks * 32 + hi * 8;
#pragma unroll
      for (int j = 0; j < 4; ++j) {
        Frag bfr;
        bfr.q[0] = *(const u32x4*)&Ks[j * 16 + l15][kbb];
        bfr.q[1] = *(const u32x4*)&Ks[j * 16 + l15][kbb + 16];
        s[j] = wmma_bf16(af.v, bfr.v, s[j]);
      }
    }

    // online softmax: rows live in 16-lane halves; reduce with shfl_xor 1..8
#pragma unroll
    for (int e = 0; e < 8; ++e) {
      float t = fmaxf(fmaxf(s[0][e], s[1][e]), fmaxf(s[2][e], s[3][e])) * scale;
#pragma unroll
      for (int msk = 1; msk < 16; msk <<= 1) t = fmaxf(t, __shfl_xor(t, msk, 32));
      const float mnew  = fmaxf(mrow[e], t);
      const float alpha = __expf(mrow[e] - mnew);
      mrow[e] = mnew;
      float psum = 0.f;
#pragma unroll
      for (int j = 0; j < 4; ++j) {
        const float p = __expf(s[j][e] * scale - mnew);
        psum += p;
        Ps[w][(hi << 3) + e][j * 16 + l15] = (bf16)p;   // wave-local LDS, in-order
      }
#pragma unroll
      for (int msk = 1; msk < 16; msk <<= 1) psum += __shfl_xor(psum, msk, 32);
      lrow[e] = lrow[e] * alpha + psum;
#pragma unroll
      for (int j = 0; j < 4; ++j) o[j][e] *= alpha;
    }

    // O += P V  (contraction over 64 keys, two K=32 steps)
#pragma unroll
    for (int ks = 0; ks < 2; ++ks) {
      Frag pf;
      const int abase = ks * 32 + hi * 8;
      pf.q[0] = *(const u32x4*)&Ps[w][l15][abase];
      pf.q[1] = *(const u32x4*)&Ps[w][l15][abase + 16];
      const int kbb = ks * 32 + hi * 8;
#pragma unroll
      for (int j = 0; j < 4; ++j) {
        Frag vf;
        vf.q[0] = *(const u32x4*)&Vts[j * 16 + l15][kbb];
        vf.q[1] = *(const u32x4*)&Vts[j * 16 + l15][kbb + 16];
        o[j] = wmma_bf16(pf.v, vf.v, o[j]);
      }
    }
  }

  // epilogue: normalize and store bf16 [B,T,H*hd]
  const int trow0 = qblk * 128 + w * 16 + hi * 8;
#pragma unroll
  for (int j = 0; j < 4; ++j) {
    const int col = h * 64 + j * 16 + l15;
#pragma unroll
    for (int e = 0; e < 8; ++e) {
      const float v = o[j][e] / lrow[e];
      attn_out[(size_t)(b * T + trow0 + e) * D + col] = (bf16)v;
    }
  }
}

// ---------------------------------------------------------------------------
extern "C" void kernel_launch(void* const* d_in, const int* in_sizes, int n_in,
                              void* d_out, int out_size, void* d_ws, size_t ws_size,
                              hipStream_t stream)
{
  const float* x      = (const float*)d_in[0];
  const float* W_qkv  = (const float*)d_in[1];
  const float* b_qkv  = (const float*)d_in[2];
  const float* W_out  = (const float*)d_in[3];
  const float* b_out  = (const float*)d_in[4];
  const float* W_gate = (const float*)d_in[5];
  const float* b_gate = (const float*)d_in[6];
  float* out = (float*)d_out;

  const int Btok = 8 * 1024;   // B*T
  const int D = 1024, D3 = 3072;

  char* ws = (char*)d_ws;
  size_t off = 0;
  auto alloc = [&](size_t bytes) -> char* {
    char* p = ws + off;
    off += (bytes + 255) & ~(size_t)255;
    return p;
  };
  bf16*  xb    = (bf16*) alloc((size_t)Btok * D  * 2);   // 16 MB
  bf16*  wqkvb = (bf16*) alloc((size_t)D    * D3 * 2);   //  6 MB
  bf16*  wgb   = (bf16*) alloc((size_t)D    * D  * 2);   //  2 MB
  bf16*  wob   = (bf16*) alloc((size_t)D    * D  * 2);   //  2 MB
  bf16*  qkvb  = (bf16*) alloc((size_t)Btok * D3 * 2);   // 48 MB
  float* gatef = (float*)alloc((size_t)Btok * D  * 4);   // 32 MB
  bf16*  attnb = (bf16*) alloc((size_t)Btok * D  * 2);   // 16 MB

  cvt_f32_bf16<<<2048, 256, 0, stream>>>(x,      xb,    Btok * D);
  cvt_f32_bf16<<<2048, 256, 0, stream>>>(W_qkv,  wqkvb, D * D3);
  cvt_f32_bf16<<<1024, 256, 0, stream>>>(W_gate, wgb,   D * D);
  cvt_f32_bf16<<<1024, 256, 0, stream>>>(W_out,  wob,   D * D);

  // gate = SiLU(x @ W_gate + b_gate), f32
  gemm_bf16_wmma<1><<<dim3(D / 128, Btok / 128), 256, 0, stream>>>(
      xb, wgb, b_gate, nullptr, gatef, nullptr, Btok, D, D);
  // qkv = x @ W_qkv + b_qkv, bf16
  gemm_bf16_wmma<0><<<dim3(D3 / 128, Btok / 128), 256, 0, stream>>>(
      xb, wqkvb, b_qkv, nullptr, nullptr, qkvb, Btok, D3, D);
  // attention -> attnb bf16
  attention_kernel<<<8 * 16 * 8, 256, 0, stream>>>(qkvb, attnb);
  // out = (attn @ W_out + b_out) * gate, f32
  gemm_bf16_wmma<2><<<dim3(D / 128, Btok / 128), 256, 0, stream>>>(
      attnb, wob, b_out, gatef, out, nullptr, Btok, D, D);
}
